// FCBlock_Mtx_77927886618978
// MI455X (gfx1250) — matrix operand order repaired and checked by reading, not compiled, not run on hardware
//
#include <hip/hip_runtime.h>
#include <hip/hip_bf16.h>

typedef __attribute__((ext_vector_type(16))) _Float16 v16h;
typedef __attribute__((ext_vector_type(8)))  _Float16 v8h;
typedef __attribute__((ext_vector_type(4)))  _Float16 v4h;
typedef __attribute__((ext_vector_type(8)))  float    v8f;
typedef __attribute__((ext_vector_type(4)))  unsigned int u32x4;
typedef __attribute__((ext_vector_type(8)))  int      i32x8;
typedef __attribute__((ext_vector_type(4)))  int      i32x4;

#define NUM_HEADS 4
#define FIN 4096
#define FOUT 4096
#define M_TOT 4096                      // 2 * 2048 rows of inp
#define HEAD_ROWS (FOUT / NUM_HEADS)    // 1024
#define KB 64                           // K-step per LDS tile (2 WMMA k-steps)

// ---- Tensor Data Mover availability / arity selection -----------------------
#if defined(__has_builtin)
# if __has_builtin(__builtin_amdgcn_tensor_load_to_lds)
#  define HAVE_TDM 1
# endif
#endif
#ifndef HAVE_TDM
# define HAVE_TDM 0
#endif
#if HAVE_TDM
# if defined(__has_include)
#  if __has_include(<hip/amd_detail/amd_gfx1250_TDM.h>)
#   define TDM_6ARG 1
#  else
#   define TDM_6ARG 0
#  endif
# else
#  define TDM_6ARG 0
# endif
#endif

// ---------------------------------------------------------------------------
// Prep 1: transform the weight (per-head 3x3 correlation + bias + sigmoid-gated
// residual), emit as f16 in [FOUT][FIN] layout (K contiguous -> B matrix rows).
// ---------------------------------------------------------------------------
__global__ void transform_weight_f16(const float* __restrict__ W,
                                     const float* __restrict__ conv_w,
                                     const float* __restrict__ conv_b,
                                     const float* __restrict__ sk_wt,
                                     _Float16* __restrict__ Bh) {
  const int idx = blockIdx.x * blockDim.x + threadIdx.x;   // one element each
  const int c = idx & (FIN - 1);
  const int o = idx >> 12;                                 // / FIN
  if (o >= FOUT) return;
  const int h = o >> 10;                                   // / HEAD_ROWS
  const int r = o & (HEAD_ROWS - 1);
  const float* cw = conv_w + h * 9;
  const float sig = 1.0f / (1.0f + __expf(-sk_wt[h]));
  float acc = conv_b[h] + sig * W[(size_t)o * FIN + c];
  #pragma unroll
  for (int dr = -1; dr <= 1; ++dr) {
    const int rr = r + dr;
    if (rr < 0 || rr >= HEAD_ROWS) continue;       // zero pad at head boundary
    #pragma unroll
    for (int dc = -1; dc <= 1; ++dc) {
      const int cc = c + dc;
      if (cc < 0 || cc >= FIN) continue;           // zero pad at column edge
      acc += cw[(dr + 1) * 3 + (dc + 1)] * W[(size_t)(h * HEAD_ROWS + rr) * FIN + cc];
    }
  }
  Bh[(size_t)o * FIN + c] = (_Float16)acc;
}

// ---------------------------------------------------------------------------
// Prep 2: cast activations f32 -> f16, [M_TOT][FIN] (K contiguous -> A rows).
// ---------------------------------------------------------------------------
__global__ void cast_inp_f16(const float* __restrict__ inp,
                             _Float16* __restrict__ Ah, int n) {
  const int i = (blockIdx.x * blockDim.x + threadIdx.x) * 4;
  if (i >= n) return;
  const float4 v = *(const float4*)(inp + i);
  v4h hv;
  hv[0] = (_Float16)v.x; hv[1] = (_Float16)v.y;
  hv[2] = (_Float16)v.z; hv[3] = (_Float16)v.w;
  *(v4h*)(Ah + i) = hv;
}

#if HAVE_TDM
// Issue one TDM descriptor: DMA a 128-row x 64-col f16 tile (row stride 4096
// elements) from global memory into LDS at lds_off. D# layout per ISA §8.
__device__ __forceinline__ void tdm_load_tile(const void* gptr, unsigned lds_off) {
  const unsigned long long ga = (unsigned long long)(size_t)gptr;
  u32x4 g0;
  g0[0] = 1u;                                   // count=1, user mode, no gather
  g0[1] = lds_off;                              // lds_addr
  g0[2] = (unsigned)ga;                         // global_addr[31:0]
  g0[3] = (unsigned)((ga >> 32) & 0x1FFFFFFull) // global_addr[56:32]
        | (2u << 30);                           // type=2 ("image")
  i32x8 g1;
  g1[0] = 0x00010000;                 // wg_mask=0, data_size=1 (2 bytes)
  g1[1] = (int)0x10000000;            // tensor_dim0[15:0]=4096 at bits[63:48]
  g1[2] = (int)0x10000000;            // dim0 hi=0 | tensor_dim1[15:0]=4096
  g1[3] = (KB << 16);                 // tensor_dim1 hi=0 | tile_dim0=64
  g1[4] = 128;                        // tile_dim1=128, tile_dim2=0
  g1[5] = 4096;                       // tensor_dim0_stride = 4096 elements
  g1[6] = 0;                          // stride hi / dim1_stride (unused, 2D)
  g1[7] = 0;
  const i32x4 z4 = {0, 0, 0, 0};
#if TDM_6ARG
  const i32x8 z8 = {0, 0, 0, 0, 0, 0, 0, 0};
  __builtin_amdgcn_tensor_load_to_lds(g0, g1, z4, z4, z8, 0);
#else
  __builtin_amdgcn_tensor_load_to_lds(g0, g1, z4, z4, 0);
#endif
}
#endif

// ---------------------------------------------------------------------------
// Main GEMM: out[M_TOT][FOUT] f32 = Ah[M][K] * Bh[N][K]^T via
// v_wmma_f32_16x16x32_f16. Block = 256 threads (8 wave32), tile 128x128,
// K-step 64 (16 WMMAs/wave per barrier), double-buffered LDS (64 KB),
// tiles staged by the Tensor Data Mover when available.
// ---------------------------------------------------------------------------
__global__ __launch_bounds__(256)
void gemm_f16_wmma(const _Float16* __restrict__ Ah,
                   const _Float16* __restrict__ Bh,
                   float* __restrict__ out) {
  __shared__ __align__(16) _Float16 Alds[2][128][KB];
  __shared__ __align__(16) _Float16 Blds[2][128][KB];

  const int tid    = threadIdx.x;
  const int lane   = tid & 31;
  const int wave   = tid >> 5;       // 0..7
  const int waveM  = wave & 3;       // 4 waves along M -> 32 rows each
  const int waveN  = wave >> 2;      // 2 waves along N -> 64 cols each
  const int blockM = blockIdx.y * 128;
  const int blockN = blockIdx.x * 128;

  v8f acc[2][4];
  #pragma unroll
  for (int i = 0; i < 2; ++i)
    #pragma unroll
    for (int j = 0; j < 4; ++j)
      #pragma unroll
      for (int v = 0; v < 8; ++v) acc[i][j][v] = 0.0f;

  // ---- tile staging: TDM descriptor issue (wave 0) or manual fallback ----
  auto stage = [&](int kb, int buf) {
#if HAVE_TDM
    if (wave == 0) {                 // wave-uniform branch; one DMA per matrix
      tdm_load_tile(Ah + (size_t)blockM * FIN + kb * KB,
                    (unsigned)(size_t)&Alds[buf][0][0]);
      tdm_load_tile(Bh + (size_t)blockN * FIN + kb * KB,
                    (unsigned)(size_t)&Blds[buf][0][0]);
    }
#else
    const int kBase = kb * KB;
    #pragma unroll
    for (int t = 0; t < 4; ++t) {
      const int ch = tid + t * 256;       // 1024 16B chunks per matrix
      const int r  = ch >> 3;             // 0..127
      const int c8 = (ch & 7) << 3;       // 0..56 step 8
      const uint4 av = *(const uint4*)(Ah + (size_t)(blockM + r) * FIN + kBase + c8);
      const uint4 bv = *(const uint4*)(Bh + (size_t)(blockN + r) * FIN + kBase + c8);
      *(uint4*)(&Alds[buf][r][c8]) = av;
      *(uint4*)(&Blds[buf][r][c8]) = bv;
    }
#endif
  };
  auto stage_fence = [&]() {
#if HAVE_TDM
    if (wave == 0) __builtin_amdgcn_s_wait_tensorcnt(0);
#endif
    __syncthreads();
  };

  const int KBLOCKS = FIN / KB;      // 64 iterations
  stage(0, 0);
  stage_fence();

  for (int kb = 0; kb < KBLOCKS; ++kb) {
    const int cur = kb & 1;
    const int nxt = cur ^ 1;
    if (kb + 1 < KBLOCKS) stage(kb + 1, nxt);   // overlaps with compute below

    const int mrow  = waveM * 32 + (lane & 15);
    const int ncol  = waveN * 64 + (lane & 15);
    const int koffA = (lane < 16) ? 0 : 8;      // A: two 8-half chunks per lane
    const int koffB = (lane < 16) ? 0 : 8;      // B: two 8-half chunks per lane

    #pragma unroll
    for (int kh = 0; kh < 2; ++kh) {            // two WMMA k-steps per tile
      v16h a[2];
      #pragma unroll
      for (int i = 0; i < 2; ++i) {
        const v8h lo = *(const v8h*)(&Alds[cur][mrow + i * 16][kh * 32 + koffA]);
        const v8h hi = *(const v8h*)(&Alds[cur][mrow + i * 16][kh * 32 + koffA + 16]);
        a[i] = __builtin_shufflevector(lo, hi, 0, 1, 2, 3, 4, 5, 6, 7,
                                               8, 9, 10, 11, 12, 13, 14, 15);
      }
      v16h b[4];
      #pragma unroll
      for (int j = 0; j < 4; ++j)
        b[j] = __builtin_shufflevector(*(const v8h*)(&Blds[cur][ncol + j * 16][kh * 32 + koffB]), *(const v8h*)(&Blds[cur][ncol + j * 16][kh * 32 + koffB + 16]), 0, 1, 2, 3, 4, 5, 6, 7, 8, 9, 10, 11, 12, 13, 14, 15);

      #pragma unroll
      for (int i = 0; i < 2; ++i)
        #pragma unroll
        for (int j = 0; j < 4; ++j)
          acc[i][j] = __builtin_amdgcn_wmma_f32_16x16x32_f16(
              false, a[i], false, b[j], (short)0, acc[i][j], false, false);
    }
    stage_fence();   // next buffer fully in LDS + this buffer's reads retired
  }

  // ---- epilogue: 16x16 f32 C layout -> VGPR v holds M=v (lanes 0-15) / M=v+8
  const int rsel = (lane >> 4) << 3;   // +8 rows for upper half-wave
  #pragma unroll
  for (int i = 0; i < 2; ++i) {
    #pragma unroll
    for (int j = 0; j < 4; ++j) {
      const int row0 = blockM + waveM * 32 + i * 16 + rsel;
      const int col  = blockN + waveN * 64 + j * 16 + (lane & 15);
      #pragma unroll
      for (int v = 0; v < 8; ++v)
        out[(size_t)(row0 + v) * FOUT + col] = acc[i][j][v];
    }
  }
}

// ---------------------------------------------------------------------------
extern "C" void kernel_launch(void* const* d_in, const int* in_sizes, int n_in,
                              void* d_out, int out_size, void* d_ws, size_t ws_size,
                              hipStream_t stream) {
  const float* inp    = (const float*)d_in[0];   // [2,2048,4096]
  const float* W      = (const float*)d_in[1];   // [4096,4096]
  const float* conv_w = (const float*)d_in[2];   // [4,1,3,3]
  const float* conv_b = (const float*)d_in[3];   // [4]
  const float* sk_wt  = (const float*)d_in[4];   // [4,1,1]
  float* out = (float*)d_out;                    // [2,2048,4096] f32

  _Float16* Ah = (_Float16*)d_ws;                        // 32 MB
  _Float16* Bh = Ah + (size_t)M_TOT * FIN;               // 32 MB

  // Prep: cast activations, transform weights (both bandwidth-bound)
  cast_inp_f16<<<(M_TOT * FIN / 4 + 255) / 256, 256, 0, stream>>>(
      inp, Ah, M_TOT * FIN);
  transform_weight_f16<<<(FOUT * FIN + 255) / 256, 256, 0, stream>>>(
      W, conv_w, conv_b, sk_wt, Bh);

  // Main compute-bound GEMM on the WMMA path
  dim3 grid(FOUT / 128, M_TOT / 128);   // 32 x 32 blocks
  gemm_f16_wmma<<<grid, 256, 0, stream>>>(Ah, Bh, out);
}
